// MyGCN_40114994545241
// MI455X (gfx1250) — matrix operand order repaired and checked by reading, not compiled, not run on hardware
//
#include <hip/hip_runtime.h>
#include <cstdint>

#define GN 2048   // N
#define GF 128    // F
#define GU 128    // U
#define TM 128    // M rows per block (main GEMM)
#define TK 32     // K step (bf16 WMMA K)
#define LDSTR 40  // halfs per LDS row: 32 data + 8 pad (bank-conflict-free b128 reads)

#define USE_ASYNC_LDS 1

typedef __attribute__((ext_vector_type(16))) __bf16 v16bf;
typedef __attribute__((ext_vector_type(8)))  float  v8f;

struct Frag32 { uint4 lo; uint4 hi; };  // 32 bytes == v16bf

// round-to-nearest (half-up) f32 -> bf16, packed pair; ~3 VALU per pair
static __device__ __forceinline__ unsigned pk2(float a, float b) {
  unsigned ra = __float_as_uint(a) + 0x8000u;
  unsigned rb = __float_as_uint(b) + 0x8000u;
  return (ra >> 16) | (rb & 0xFFFF0000u);
}

// ---------------- Kernel 1a: partial column sums of A (+ optional A->bf16 preconvert) -------
// grid (GN/1024, 16 slabs of 128 rows, B), block 256; float4 streaming.
template <bool WBF>
__global__ __launch_bounds__(256) void colsum_kernel(const float* __restrict__ A,
                                                     float* __restrict__ partial,
                                                     unsigned short* __restrict__ Abf) {
  const int b    = blockIdx.z;
  const int slab = blockIdx.y;
  const int col  = blockIdx.x * 1024 + threadIdx.x * 4;
  const float* base = A + (size_t)b * GN * GN + (size_t)slab * 128 * GN + col;
  float4 acc = {0.f, 0.f, 0.f, 0.f};
  for (int r = 0; r < 128; ++r) {
    float4 v = *(const float4*)(base + (size_t)r * GN);
    if (WBF) {
      uint2 pv;
      pv.x = pk2(v.x, v.y);
      pv.y = pk2(v.z, v.w);
      *(uint2*)(Abf + ((size_t)b * GN + slab * 128 + r) * GN + col) = pv;
    }
    acc.x += v.x; acc.y += v.y; acc.z += v.z; acc.w += v.w;
  }
  *(float4*)(partial + ((size_t)b * 16 + slab) * GN + col) = acc;
}

// ---------------- Kernel 1b: reduce partials -> kc = k^-0.25 ----------------
__global__ __launch_bounds__(256) void kc_kernel(const float* __restrict__ partial,
                                                 const float* __restrict__ pin,
                                                 float* __restrict__ kc) {
  const int i = blockIdx.x * 256 + threadIdx.x;  // i = b*GN + n
  const int b = i >> 11;
  const int n = i & (GN - 1);
  float s = 0.f;
#pragma unroll
  for (int j = 0; j < 16; ++j) s += partial[((size_t)b * 16 + j) * GN + n];
  const float ps  = 1.0f / (1.0f + expf(-pin[0]));
  const float k   = ps + (1.0f - ps) * s;
  const float kir = 1.0f / sqrtf(sqrtf(k));          // k^-0.25
  kc[i] = isinf(kir) ? 0.0f : kir;
}

// ---------------- Kernel 2: Y = kc * (X @ W); store Yf (f32) + Yt (bf16, transposed) --------
__global__ __launch_bounds__(256) void gcn_xw_kernel(const float* __restrict__ X,
                                                     const float* __restrict__ W,
                                                     const float* __restrict__ kc,
                                                     float* __restrict__ Yf,
                                                     unsigned short* __restrict__ Yt) {
  __shared__ __align__(16) float Xs[16 * GF];
  const int b  = blockIdx.y;
  const int n0 = blockIdx.x * 16;
  const int t  = threadIdx.x;
  const float4* xsrc = (const float4*)(X + ((size_t)b * GN + n0) * GF);
  ((float4*)Xs)[t]       = xsrc[t];
  ((float4*)Xs)[t + 256] = xsrc[t + 256];
  __syncthreads();

  const int u    = t & (GU - 1);
  const int half = t >> 7;      // 0/1: rows [half*8, half*8+8)
  float facc[8] = {};
  for (int f = 0; f < GF; ++f) {
    const float w = W[(size_t)f * GU + u];   // coalesced across u
#pragma unroll
    for (int r = 0; r < 8; ++r)
      facc[r] = fmaf(Xs[(half * 8 + r) * GF + f], w, facc[r]);  // LDS broadcast
  }
  float y[8];
#pragma unroll
  for (int r = 0; r < 8; ++r) {
    const int n = n0 + half * 8 + r;
    const float kcn = kc[(size_t)b * GN + n];
    y[r] = kcn * facc[r];
    Yf[((size_t)b * GN + n) * GU + u] = y[r];
  }
  uint4 pkv;
  pkv.x = pk2(y[0], y[1]); pkv.y = pk2(y[2], y[3]);
  pkv.z = pk2(y[4], y[5]); pkv.w = pk2(y[6], y[7]);
  // transposed bf16: Yt[b][u][n0 + half*8 .. +8)  (one aligned 16B store)
  *(uint4*)(Yt + ((size_t)b * GU + u) * GN + n0 + half * 8) = pkv;
}

// ---------------- Kernel 3: out = relu(kc[n]*(A@Y) + qs*kc[n]*Y + bias) ----------------
// grid (GN/TM, B), block 256 (8 wave32). BF16 WMMA 16x16x32, double-buffered LDS.
// PRECONV=true: both operands staged with async global->LDS DMA (A_bf16 from ws).
// PRECONV=false: A converted f32->bf16 in-kernel (fallback when ws is small).
template <bool PRECONV>
__global__ __launch_bounds__(256) void gcn_main_kernel(const float* __restrict__ A,
                                                       const unsigned short* __restrict__ Abf,
                                                       const float* __restrict__ Yf,
                                                       const unsigned short* __restrict__ Yt,
                                                       const float* __restrict__ kc,
                                                       const float* __restrict__ bias,
                                                       const float* __restrict__ qin,
                                                       float* __restrict__ out) {
  __shared__ __align__(16) unsigned short As[2][TM * LDSTR];  // A tile, bf16, padded rows
  __shared__ __align__(16) unsigned short Bs[2][GU * LDSTR];  // Yt tile, bf16, padded rows

  const int b    = blockIdx.y;
  const int m0   = blockIdx.x * TM;
  const int tid  = threadIdx.x;
  const int wv   = tid >> 5;
  const int lane = tid & 31;
  const int hi   = lane >> 4;
  const int lm   = lane & 15;

  const int srow  = tid >> 1;  // 0..127 : tile row handled by this thread when staging
  const int shalf = tid & 1;   // 0/1    : which 16-half chunk of the 32-wide row

  const float* Ablk = A + ((size_t)b * GN + m0) * GN;
  const unsigned short* Abblk = Abf + ((size_t)b * GN + m0) * GN;
  const unsigned short* Ytb = Yt + (size_t)b * GU * GN;

  v8f acc[8] = {};

  auto asyncCopy32 = [&](unsigned short* dst, const unsigned short* src) {
#if USE_ASYNC_LDS
    unsigned lofs = (unsigned)(size_t)dst;  // flat LDS ptr low 32 bits == LDS offset
    asm volatile("global_load_async_to_lds_b128 %0, %1, off"
                 :: "v"(lofs), "v"(src) : "memory");
    asm volatile("global_load_async_to_lds_b128 %0, %1, off offset:16"
                 :: "v"(lofs), "v"(src) : "memory");
#else
    const uint4* s4 = (const uint4*)src;
    uint4* d4 = (uint4*)dst;
    d4[0] = s4[0]; d4[1] = s4[1];
#endif
  };

  auto stageA = [&](int kk, int buf) {
    unsigned short* dst = &As[buf][srow * LDSTR + shalf * 16];
    if (PRECONV) {
      asyncCopy32(dst, Abblk + (size_t)srow * GN + kk * TK + shalf * 16);
    } else {  // global f32 -> bf16 -> LDS
      const float4* s4 = (const float4*)(Ablk + (size_t)srow * GN + kk * TK + shalf * 16);
      float4 f0 = s4[0], f1 = s4[1], f2 = s4[2], f3 = s4[3];
      uint4 p0, p1;
      p0.x = pk2(f0.x, f0.y); p0.y = pk2(f0.z, f0.w);
      p0.z = pk2(f1.x, f1.y); p0.w = pk2(f1.z, f1.w);
      p1.x = pk2(f2.x, f2.y); p1.y = pk2(f2.z, f2.w);
      p1.z = pk2(f3.x, f3.y); p1.w = pk2(f3.z, f3.w);
      uint4* d4 = (uint4*)dst;
      d4[0] = p0; d4[1] = p1;
    }
  };

  auto stageB = [&](int kk, int buf) {
    asyncCopy32(&Bs[buf][srow * LDSTR + shalf * 16],
                Ytb + (size_t)srow * GN + kk * TK + shalf * 16);
  };

  auto waitAsync = [&]() {
#if USE_ASYNC_LDS
#if __has_builtin(__builtin_amdgcn_s_wait_asynccnt)
    __builtin_amdgcn_s_wait_asynccnt(0);
#else
    asm volatile("s_wait_asynccnt 0" ::: "memory");
#endif
#endif
  };

  auto compute = [&](int buf) {
    // A fragment: row = wv*16+lm ; K halfs [hi*16,+16) contiguous (matches B)
    const unsigned short* ab = &As[buf][(wv * 16 + lm) * LDSTR];
    Frag32 fa;
    fa.lo = *(const uint4*)(ab + hi * 16);
    fa.hi = *(const uint4*)(ab + hi * 16 + 8);
    v16bf a = __builtin_bit_cast(v16bf, fa);
#pragma unroll
    for (int nt = 0; nt < 8; ++nt) {
      // B fragment: col u = nt*16+lm ; K halfs [hi*16,+16) contiguous in transposed tile
      const unsigned short* bb = &Bs[buf][(nt * 16 + lm) * LDSTR + hi * 16];
      Frag32 fb;
      fb.lo = ((const uint4*)bb)[0];
      fb.hi = ((const uint4*)bb)[1];
      v16bf bv = __builtin_bit_cast(v16bf, fb);
      acc[nt] = __builtin_amdgcn_wmma_f32_16x16x32_bf16(
          false, a, false, bv, (short)0, acc[nt], false, false);
    }
  };

  stageB(0, 0);
  stageA(0, 0);
  waitAsync();
  __syncthreads();

  int buf = 0;
  const int NK = GN / TK;  // 64
  for (int kk = 0; kk < NK; ++kk) {
    if (kk + 1 < NK) { stageB(kk + 1, buf ^ 1); stageA(kk + 1, buf ^ 1); }
    compute(buf);
    waitAsync();
    __syncthreads();
    buf ^= 1;
  }

  // Epilogue: kc row scale + q-diagonal term + bias + relu.
  const float qs = 1.0f / (1.0f + expf(-qin[0]));
  const float* kcb = kc + (size_t)b * GN;
  const float* Yfb = Yf + (size_t)b * GN * GU;
#pragma unroll
  for (int nt = 0; nt < 8; ++nt) {
    const int u = nt * 16 + lm;
    const float bu = bias[u];
#pragma unroll
    for (int r = 0; r < 8; ++r) {
      const int n = m0 + wv * 16 + r + 8 * hi;      // C/D layout: VGPR r, lane-half hi
      const float kcn = kcb[n];
      const float yv  = Yfb[(size_t)n * GU + u];
      const float v   = fmaf(kcn, acc[nt][r], fmaf(qs * kcn, yv, bu));
      out[((size_t)b * GN + n) * GU + u] = fmaxf(v, 0.0f);
    }
  }
}

// ---------------- launcher ----------------
extern "C" void kernel_launch(void* const* d_in, const int* in_sizes, int n_in,
                              void* d_out, int out_size, void* d_ws, size_t ws_size,
                              hipStream_t stream) {
  const float* A    = (const float*)d_in[0];
  const float* X    = (const float*)d_in[1];
  const float* W    = (const float*)d_in[2];
  const float* bias = (const float*)d_in[3];
  const float* p    = (const float*)d_in[4];
  const float* q    = (const float*)d_in[5];
  float* out = (float*)d_out;
  const int B = in_sizes[0] / (GN * GN);  // 8

  // workspace layout: kc | partial | Yf | Yt | (optional) Abf
  float* kc      = (float*)d_ws;                                    // B*GN f32
  float* partial = kc + (size_t)B * GN;                             // B*16*GN f32
  float* Yf      = partial + (size_t)B * 16 * GN;                   // B*GN*GU f32
  unsigned short* Yt  = (unsigned short*)(Yf + (size_t)B * GN * GU);// B*GU*GN bf16
  unsigned short* Abf = Yt + (size_t)B * GU * GN;                   // B*GN*GN bf16

  const size_t need_base = ((size_t)B * GN + (size_t)B * 16 * GN +
                            (size_t)B * GN * GU) * sizeof(float) +
                           (size_t)B * GU * GN * sizeof(unsigned short);
  const size_t need_abf  = (size_t)B * GN * GN * sizeof(unsigned short);
  const bool preconv = (ws_size >= need_base + need_abf);

  if (preconv) {
    colsum_kernel<true><<<dim3(GN / 1024, 16, B), 256, 0, stream>>>(A, partial, Abf);
  } else {
    colsum_kernel<false><<<dim3(GN / 1024, 16, B), 256, 0, stream>>>(A, partial, nullptr);
  }
  kc_kernel<<<dim3((B * GN) / 256), 256, 0, stream>>>(partial, p, kc);
  gcn_xw_kernel<<<dim3(GN / 16, B), 256, 0, stream>>>(X, W, kc, Yf, Yt);
  if (preconv) {
    gcn_main_kernel<true><<<dim3(GN / TM, B), 256, 0, stream>>>(A, Abf, Yf, Yt, kc, bias, q, out);
  } else {
    gcn_main_kernel<false><<<dim3(GN / TM, B), 256, 0, stream>>>(A, nullptr, Yf, Yt, kc, bias, q, out);
  }
}
